// SelfAttention_33706903339388
// MI455X (gfx1250) — matrix operand order repaired and checked by reading, not compiled, not run on hardware
//
#include <hip/hip_runtime.h>
#include <stdint.h>
#include <stddef.h>

// ---------------------------------------------------------------------------
// MI455X (gfx1250) implementation of SAGAN self-attention w/ spectral norm.
// All GEMMs use v_wmma_f32_16x16x32_bf16 (fp32 accumulate). Attention is
// fused flash-style (no beta materialization: saves 2x256MB of HBM traffic,
// ~22us at 23.3 TB/s). Intermediates stored bf16 in layouts matching the
// CDNA5 WMMA A/B VGPR striping so operand loads are contiguous b128s.
// K/V chunk staging uses GLOBAL_LOAD_ASYNC_TO_LDS_B128 (ASYNCcnt) so the
// DMA writes LDS directly without a VGPR round-trip.
// ---------------------------------------------------------------------------

typedef __bf16 bf16_t;
typedef __attribute__((ext_vector_type(16))) __bf16 v16bf;
typedef __attribute__((ext_vector_type(8)))  __bf16 v8bf;
typedef __attribute__((ext_vector_type(8)))  float  v8f;

#define WMMA_BF16(a, b, c) \
  __builtin_amdgcn_wmma_f32_16x16x32_bf16(false, (a), false, (b), (short)0, (c), false, false)

static constexpr int B_  = 16;
static constexpr int C_  = 512;
static constexpr int HW  = 4096;   // 64*64
static constexpr int HWP = 1024;   // pooled 32*32
static constexpr int CQ  = 64;     // C/8
static constexpr int CV  = 256;    // C/2
static constexpr int CM  = 384;    // CQ + CQ + CV combined qkv rows

// workspace byte offsets
static constexpr size_t WBF_OFF = 0;                                   // 327680 bf16 weights
static constexpr size_t QKV_OFF = (size_t)1 << 20;                     // [B][4096][384] bf16
static constexpr size_t G_OFF   = QKV_OFF + (size_t)B_ * HW  * CM * 2; // [B][1024][64]  bf16
static constexpr size_t HV_OFF  = G_OFF   + (size_t)B_ * HWP * CQ * 2; // [B][256][1024] bf16
static constexpr size_t O_OFF   = HV_OFF  + (size_t)B_ * CV  * HWP* 2; // [B][4096][256] bf16
// weight element offsets inside WBF region
static constexpr size_t WO_E = 196608;  // after q(32768)+k(32768)+v(131072)

// Build a 16-bit WMMA A/B operand from two contiguous 8-element runs.
// A layout: lane L holds M=L%16, K runs at kbase=(L/16)*8 and kbase+16.
// B layout: lane L holds N=L%16, K runs at kbase=(L/16)*8 and kbase+16.
__device__ __forceinline__ v16bf load16_split(const bf16_t* p0, const bf16_t* p1) {
  v8bf lo = *(const v8bf*)p0;
  v8bf hi = *(const v8bf*)p1;
  v16bf r;
#pragma unroll
  for (int i = 0; i < 8; ++i) { r[i] = lo[i]; r[i + 8] = hi[i]; }
  return r;
}

// Async DMA of 16 bytes global -> LDS (no VGPR round trip, tracked by
// ASYNCcnt). LDS offset = low 32 bits of the generic shared-memory pointer
// (flat->LDS mapping truncates to addr[31:0] per CDNA5 ISA 10.2).
__device__ __forceinline__ void async_copy_b128(void* lds_dst, const void* gsrc) {
  const unsigned loff = (unsigned)(uintptr_t)lds_dst;
  const unsigned long long ga = (unsigned long long)(uintptr_t)gsrc;
  asm volatile("global_load_async_to_lds_b128 %0, %1, off"
               :
               : "v"(loff), "v"(ga)
               : "memory");
}
__device__ __forceinline__ void async_fence() {
  asm volatile("s_wait_asynccnt 0" ::: "memory");
}

// ---------------------------------------------------------------------------
// Kernel 1: spectral normalization (power iteration, 5 iters) of the 4
// weights; writes normalized bf16 copies to workspace. One block per weight.
// ---------------------------------------------------------------------------
__global__ __launch_bounds__(256) void spectral_norm_kernel(
    const float* __restrict__ w0, const float* __restrict__ w1,
    const float* __restrict__ w2, const float* __restrict__ w3,
    bf16_t* __restrict__ wbf) {
  __shared__ float u_s[512], v_s[512], red[256];
  const int which = blockIdx.x;
  const float* w = (which == 0) ? w0 : (which == 1) ? w1 : (which == 2) ? w2 : w3;
  const int M = (which < 2) ? 64 : (which == 2) ? 256 : 512;
  const int K = (which == 3) ? 256 : 512;
  const size_t off = (which == 0) ? 0 : (which == 1) ? 32768 : (which == 2) ? 65536 : WO_E;
  const int t = threadIdx.x;

  const float inv_sqrt_m = rsqrtf((float)M);
  for (int m = t; m < M; m += 256) u_s[m] = inv_sqrt_m;
  __syncthreads();

  float sigma = 1.0f;
  for (int it = 0; it < 5; ++it) {
    // v = normalize(W^T u)
    for (int k = t; k < K; k += 256) {
      float s = 0.0f;
      for (int m = 0; m < M; ++m) s += w[(size_t)m * K + k] * u_s[m];
      v_s[k] = s;
    }
    __syncthreads();
    float ls = 0.0f;
    for (int k = t; k < K; k += 256) ls += v_s[k] * v_s[k];
    red[t] = ls; __syncthreads();
    for (int s2 = 128; s2 > 0; s2 >>= 1) { if (t < s2) red[t] += red[t + s2]; __syncthreads(); }
    const float nv = sqrtf(red[0]) + 1e-12f;
    __syncthreads();
    for (int k = t; k < K; k += 256) v_s[k] /= nv;
    __syncthreads();
    // u = normalize(W v); sigma = ||W v|| on last iter (== u . (W v))
    for (int m = t; m < M; m += 256) {
      float s = 0.0f;
      for (int k = 0; k < K; ++k) s += w[(size_t)m * K + k] * v_s[k];
      u_s[m] = s;
    }
    __syncthreads();
    ls = 0.0f;
    for (int m = t; m < M; m += 256) ls += u_s[m] * u_s[m];
    red[t] = ls; __syncthreads();
    for (int s2 = 128; s2 > 0; s2 >>= 1) { if (t < s2) red[t] += red[t + s2]; __syncthreads(); }
    const float nu = sqrtf(red[0]);
    sigma = nu;
    __syncthreads();
    const float nud = nu + 1e-12f;
    for (int m = t; m < M; m += 256) u_s[m] /= nud;
    __syncthreads();
  }
  const float inv_sigma = 1.0f / sigma;
  const size_t total = (size_t)M * K;
  for (size_t i = t; i < total; i += 256) wbf[off + i] = (bf16_t)(w[i] * inv_sigma);
}

// ---------------------------------------------------------------------------
// Kernel 2: fused q/k/v 1x1 conv as one WMMA GEMM: [384x512] x [512 x N].
// Each wave owns one 16x16 output tile; output stored pixel-major bf16
// ([b][n][384]) so later kernels read contiguous channel runs.
// ---------------------------------------------------------------------------
__global__ __launch_bounds__(256) void qkv_gemm_kernel(
    const float* __restrict__ x, const bf16_t* __restrict__ wbf,
    bf16_t* __restrict__ qkv) {
  const int lane = threadIdx.x & 31;
  const int nl = lane & 15, half = lane >> 4;
  const int flat = blockIdx.x * 8 + (threadIdx.x >> 5);
  const int mt = flat % 24;
  const int nt = (flat / 24) & 255;
  const int bb = flat / (24 * 256);

  const bf16_t* wrow = wbf + (size_t)(mt * 16 + nl) * 512;        // A: weight row M
  const float*  xcol = x + (size_t)bb * C_ * HW + (nt * 16 + nl); // B: pixel column N

  v8f acc = {0.f,0.f,0.f,0.f,0.f,0.f,0.f,0.f};
#pragma unroll 4
  for (int k0 = 0; k0 < 512; k0 += 32) {
    v16bf av = load16_split(wrow + k0 + half * 8, wrow + k0 + half * 8 + 16);
    v16bf bv;
#pragma unroll
    for (int j = 0; j < 16; ++j)
      bv[j] = (bf16_t)xcol[(size_t)(k0 + half * 8 + (j & 7) + ((j >> 3) << 4)) * HW];   // strided channel reads
    acc = WMMA_BF16(av, bv, acc);
  }
  // C layout: lane holds 8 contiguous output channels for pixel n -> one b128 store
  bf16_t* dst = qkv + ((size_t)bb * HW + nt * 16 + nl) * CM + mt * 16 + half * 8;
  v8bf o;
#pragma unroll
  for (int r = 0; r < 8; ++r) o[r] = (bf16_t)acc[r];
  *(v8bf*)dst = o;
}

// ---------------------------------------------------------------------------
// Kernel 3: 2x2 maxpool of the k/v parts of qkv into
//   G: [b][m][64]  (key-pixel-major -> WMMA A operand rows for scores)
//   H: [b][c][1024] (channel-major -> WMMA A operand rows for P.V)
// ---------------------------------------------------------------------------
__global__ __launch_bounds__(320) void pool_kernel(
    const bf16_t* __restrict__ qkv, bf16_t* __restrict__ gk, bf16_t* __restrict__ hv) {
  const int m  = blockIdx.x & 1023;
  const int bb = blockIdx.x >> 10;
  const int hp = m >> 5, wp = m & 31;
  const int n00 = hp * 2 * 64 + wp * 2;
  const int t = threadIdx.x;            // 0..319 -> channels 64..383 of qkv
  const bf16_t* q = qkv + (size_t)bb * HW * CM;
  const int cc = 64 + t;
  const float v0 = (float)q[(size_t)n00 * CM + cc];
  const float v1 = (float)q[(size_t)(n00 + 1) * CM + cc];
  const float v2 = (float)q[(size_t)(n00 + 64) * CM + cc];
  const float v3 = (float)q[(size_t)(n00 + 65) * CM + cc];
  const float mx = fmaxf(fmaxf(v0, v1), fmaxf(v2, v3));
  if (t < 64) gk[((size_t)bb * HWP + m) * CQ + t] = (bf16_t)mx;
  else        hv[((size_t)bb * CV + (t - 64)) * HWP + m] = (bf16_t)mx;
}

// ---------------------------------------------------------------------------
// Kernel 4: fused flash attention. Block = 8 waves, each wave owns 16 queries.
// kv streamed in 64-position chunks through LDS (48KB static) via async
// global->LDS DMA (ASYNCcnt). S is computed transposed
// (S[m,n] = sum_c G[m,c]*F[n,c]) so softmax over m is per-lane across
// accumulator elements plus one shfl_xor(16) half-exchange.
// P is re-laid into WMMA B format via a tiny per-wave LDS tile.
// ---------------------------------------------------------------------------
__global__ __launch_bounds__(256) void attention_kernel(
    const bf16_t* __restrict__ qkv, const bf16_t* __restrict__ gk,
    const bf16_t* __restrict__ hv, bf16_t* __restrict__ obuf) {
  __shared__ __align__(16) bf16_t g_s[64][64];      // keys chunk   [m][c]  8 KB
  __shared__ __align__(16) bf16_t h_s[256][64];     // values chunk [c][m] 32 KB
  __shared__ __align__(16) bf16_t p_s[8][32 * 16];  // per-wave P   [m][n]  8 KB

  const int tid  = threadIdx.x;
  const int lane = tid & 31;
  const int w    = tid >> 5;
  const int nl   = lane & 15;
  const int half = lane >> 4;
  const int bb   = blockIdx.x >> 5;
  const int qt   = blockIdx.x & 31;
  const int n0   = (qt * 8 + w) * 16;

  // Preload query B-operands (channels 0..63 live at start of qkv pixel row).
  const bf16_t* frow = qkv + ((size_t)bb * HW + n0 + nl) * CM;
  const v16bf fb0 = load16_split(frow + half * 8,      frow + half * 8 + 16);
  const v16bf fb1 = load16_split(frow + 32 + half * 8, frow + 32 + half * 8 + 16);

  const v8f z8 = {0.f,0.f,0.f,0.f,0.f,0.f,0.f,0.f};
  v8f acc[16];                                   // O tile: 256 ch x 16 q (128 VGPRs)
#pragma unroll
  for (int i = 0; i < 16; ++i) acc[i] = z8;
  float mi = -3.0e38f, li = 0.0f;

  const bf16_t* gsrc = gk + (size_t)bb * HWP * CQ;   // [1024][64] chunk-contiguous

  for (int ch = 0; ch < 16; ++ch) {
    __syncthreads();
    // stage keys chunk: 64 m x 64 c = 8KB -> 2 async b128 per thread
    {
      const bf16_t* gc = gsrc + ch * (64 * 64);
      async_copy_b128(&g_s[0][0] + tid * 8,         gc + tid * 8);
      async_copy_b128(&g_s[0][0] + (tid + 256) * 8, gc + (tid + 256) * 8);
    }
    // stage values chunk: one 128B row per thread -> 8 async b128
    const bf16_t* hrow_g = hv + ((size_t)bb * CV + tid) * HWP + ch * 64;
#pragma unroll
    for (int i = 0; i < 8; ++i)
      async_copy_b128(&h_s[tid][i * 8], hrow_g + i * 8);
    if (ch + 1 < 16) __builtin_prefetch((const void*)(hrow_g + 64), 0, 1);
    async_fence();        // wave's async DMAs complete (compiler can't track asm)
    __syncthreads();      // cross-wave visibility of the staged chunk

#pragma unroll
    for (int dt = 0; dt < 2; ++dt) {
      const int mlo = dt * 32;
      // ---- scores: two 16x16 S tiles over K=64 channels (4 WMMAs) ----
      v8f s0 = z8, s1 = z8;
      {
        const bf16_t* gr0 = &g_s[mlo + nl][0];
        const bf16_t* gr1 = &g_s[mlo + 16 + nl][0];
        v16bf a;
        a = load16_split(gr0 + half * 8,      gr0 + half * 8 + 16);
        s0 = WMMA_BF16(a, fb0, s0);
        a = load16_split(gr0 + 32 + half * 8, gr0 + 32 + half * 8 + 16);
        s0 = WMMA_BF16(a, fb1, s0);
        a = load16_split(gr1 + half * 8,      gr1 + half * 8 + 16);
        s1 = WMMA_BF16(a, fb0, s1);
        a = load16_split(gr1 + 32 + half * 8, gr1 + 32 + half * 8 + 16);
        s1 = WMMA_BF16(a, fb1, s1);
      }
      // ---- online softmax over m (per query column n = nl) ----
      float vmax = -3.0e38f;
#pragma unroll
      for (int r = 0; r < 8; ++r) vmax = fmaxf(vmax, fmaxf(s0[r], s1[r]));
      vmax = fmaxf(vmax, __shfl_xor(vmax, 16, 32));   // combine the two row-halves
      const float mnew = fmaxf(mi, vmax);
      const float corr = __expf(mi - mnew);
      float e0[8], e1[8], vs = 0.0f;
#pragma unroll
      for (int r = 0; r < 8; ++r) {
        e0[r] = __expf(s0[r] - mnew);
        e1[r] = __expf(s1[r] - mnew);
        vs += e0[r] + e1[r];
      }
      vs += __shfl_xor(vs, 16, 32);
      li = li * corr + vs;
      mi = mnew;
#pragma unroll
      for (int ct = 0; ct < 16; ++ct) acc[ct] *= corr;  // rescale O accumulators

      // ---- P -> per-wave LDS tile [32m x 16n], reload in WMMA B layout ----
#pragma unroll
      for (int r = 0; r < 8; ++r) {
        p_s[w][(half * 8 + r) * 16 + nl]        = (bf16_t)e0[r];
        p_s[w][(16 + half * 8 + r) * 16 + nl]   = (bf16_t)e1[r];
      }
      asm volatile("s_wait_dscnt 0" ::: "memory");      // same-wave LDS RAW
      v16bf pb;
#pragma unroll
      for (int i = 0; i < 16; ++i) pb[i] = p_s[w][(half * 8 + (i & 7) + ((i >> 3) << 4)) * 16 + nl];

      // ---- O += H * P : 16 WMMAs over the 256 value channels ----
#pragma unroll
      for (int ct = 0; ct < 16; ++ct) {
        const bf16_t* hr = &h_s[ct * 16 + nl][0];
        v16bf ha = load16_split(hr + mlo + half * 8, hr + mlo + half * 8 + 16);
        acc[ct] = WMMA_BF16(ha, pb, acc[ct]);
      }
    }
  }

  // epilogue: normalize by softmax denominator, store pixel-major bf16
  const float inv_l = 1.0f / li;
  bf16_t* orow = obuf + ((size_t)bb * HW + n0 + nl) * CV;
#pragma unroll
  for (int ct = 0; ct < 16; ++ct) {
    v8bf o;
#pragma unroll
    for (int r = 0; r < 8; ++r) o[r] = (bf16_t)(acc[ct][r] * inv_l);
    *(v8bf*)(orow + ct * 16 + half * 8) = o;
  }
}

// ---------------------------------------------------------------------------
// Kernel 5: output 1x1 conv (Wo [512x256] GEMM) + residual: out = g*o + x.
// ---------------------------------------------------------------------------
__global__ __launch_bounds__(256) void out_gemm_kernel(
    const bf16_t* __restrict__ wo, const bf16_t* __restrict__ obuf,
    const float* __restrict__ x, const float* __restrict__ gamma_p,
    float* __restrict__ out) {
  const int lane = threadIdx.x & 31;
  const int nl = lane & 15, half = lane >> 4;
  const int flat = blockIdx.x * 8 + (threadIdx.x >> 5);
  const int mt = flat & 31;
  const int nt = (flat >> 5) & 255;
  const int bb = flat >> 13;

  const bf16_t* wrow = wo + (size_t)(mt * 16 + nl) * 256;
  const bf16_t* orow = obuf + ((size_t)bb * HW + nt * 16 + nl) * CV;

  v8f acc = {0.f,0.f,0.f,0.f,0.f,0.f,0.f,0.f};
#pragma unroll
  for (int k0 = 0; k0 < 256; k0 += 32) {
    v16bf av = load16_split(wrow + k0 + half * 8,  wrow + k0 + half * 8 + 16);
    v16bf bv = load16_split(orow + k0 + half * 8, orow + k0 + half * 8 + 16);
    acc = WMMA_BF16(av, bv, acc);
  }
  const float gamma = gamma_p[0];
  const int n = nt * 16 + nl;
  const size_t base = ((size_t)bb * C_ + mt * 16 + half * 8) * HW + n;
#pragma unroll
  for (int r = 0; r < 8; ++r) {
    const size_t idx = base + (size_t)r * HW;
    out[idx] = gamma * acc[r] + x[idx];
  }
}

// ---------------------------------------------------------------------------
extern "C" void kernel_launch(void* const* d_in, const int* in_sizes, int n_in,
                              void* d_out, int out_size, void* d_ws, size_t ws_size,
                              hipStream_t stream) {
  const float* x     = (const float*)d_in[0];
  const float* Wq    = (const float*)d_in[1];
  const float* Wk    = (const float*)d_in[2];
  const float* Wv    = (const float*)d_in[3];
  const float* Wo    = (const float*)d_in[4];
  const float* gamma = (const float*)d_in[5];
  float* out = (float*)d_out;

  char* ws = (char*)d_ws;
  bf16_t* wbf = (bf16_t*)(ws + WBF_OFF);
  bf16_t* qkv = (bf16_t*)(ws + QKV_OFF);
  bf16_t* gk  = (bf16_t*)(ws + G_OFF);
  bf16_t* hv  = (bf16_t*)(ws + HV_OFF);
  bf16_t* ob  = (bf16_t*)(ws + O_OFF);

  spectral_norm_kernel<<<4, 256, 0, stream>>>(Wq, Wk, Wv, Wo, wbf);
  qkv_gemm_kernel<<<(24 * 256 * 16) / 8, 256, 0, stream>>>(x, wbf, qkv);
  pool_kernel<<<16 * 1024, 320, 0, stream>>>(qkv, gk, hv);
  attention_kernel<<<16 * 32, 256, 0, stream>>>(qkv, gk, hv, ob);
  out_gemm_kernel<<<(32 * 256 * 16) / 8, 256, 0, stream>>>(wbf + WO_E, ob, x, gamma, out);
}
